// GGNN_13898514170322
// MI455X (gfx1250) — matrix operand order repaired and checked by reading, not compiled, not run on hardware
//
#include <hip/hip_runtime.h>
#include <hip/hip_bf16.h>
#include <math.h>

typedef __attribute__((ext_vector_type(16))) _Float16 v16h;
typedef __attribute__((ext_vector_type(8)))  _Float16 v8h;
typedef __attribute__((ext_vector_type(8)))  float    v8f;

#define HDIM  128
#define EFEAT 256

// ---------------- math helpers (branch-free, fast HW transcendentals) ----------------
// softplus(x) = max(x,0) + log(1 + exp(-|x|));  exp(-|x|) <= 1 so no overflow,
// lowers to v_exp_f32 / v_log_f32 with no divergent slow path.
__device__ __forceinline__ float sp(float v) {
  return fmaxf(v, 0.f) + __logf(1.f + __expf(-fabsf(v)));
}
__device__ __forceinline__ float sigm(float v) {
  return __builtin_amdgcn_rcpf(1.f + __expf(-v));
}

// ---------------- WMMA helpers ----------------
// A-fragment (16x32 f16, M=row striped): lane l holds row (l&15);
// lanes 0-15 hold K 0-7 (V0..3) and 16-23 (V4..7); lanes 16-31 hold K 8-15 / 24-31.
__device__ __forceinline__ v16h load_a_frag(const _Float16* rowp, int off) {
  v8h lo = *(const v8h*)(rowp + off);        // K = off .. off+7
  v8h hi = *(const v8h*)(rowp + off + 16);   // K = off+16 .. off+23
  return __builtin_shufflevector(lo, hi, 0,1,2,3,4,5,6,7,8,9,10,11,12,13,14,15);
}

__device__ __forceinline__ v8f wmma_f16(v16h a, v16h b, v8f c) {
  return __builtin_amdgcn_wmma_f32_16x16x32_f16(false, a, false, b, (short)0, c,
                                                false, false);
}

// ---------------- weight packing: f32 [K,N] -> f16 B-fragment tiles ----------------
// Tile (kt,nt) = 32x16, stored as [lane 0..31][16 f16]:
//   n = nt*16 + (lane&15) ;  k = kt*32 + ((lane&16)?8:0) + (e&7) + ((e&8)?16:0)  (e = 0..15)
__global__ __launch_bounds__(256) void pack_b(const float* __restrict__ W,
                                              _Float16* __restrict__ P, int K, int N) {
  int idx = blockIdx.x * 256 + threadIdx.x;
  if (idx >= K * N) return;
  int tile   = idx >> 9;          // 512 f16 per tile
  int within = idx & 511;
  int lane   = within >> 4;
  int e      = within & 15;
  int NT = N >> 4;
  int kt = tile / NT, nt = tile - kt * NT;
  int n = nt * 16 + (lane & 15);
  int k = kt * 32 + ((lane & 16) ? 8 : 0) + (e & 7) + ((e & 8) ? 16 : 0);
  P[idx] = (_Float16)W[(size_t)k * N + n];
}

// ---------------- embedding gather: xh[n,:] = (f16) embed[nodes[n],:] ----------------
__global__ __launch_bounds__(256) void embed_gather(const int* __restrict__ nodes,
                                                    const float* __restrict__ emb,
                                                    _Float16* __restrict__ xh, int nNodes) {
  int i = blockIdx.x * 256 + threadIdx.x;
  if (i >= nNodes * HDIM) return;
  int n = i >> 7, h = i & (HDIM - 1);
  xh[i] = (_Float16)emb[(size_t)nodes[n] * HDIM + h];
}

// ---------------- residual: xh = (f16)( xh + softplus(agg) ) ----------------
__global__ __launch_bounds__(256) void residual(_Float16* __restrict__ xh,
                                                const float* __restrict__ agg, int total) {
  int i = blockIdx.x * 256 + threadIdx.x;
  if (i < total) xh[i] = (_Float16)((float)xh[i] + sp(agg[i]));
}

// ---------------- edge conv: f16 gather -> 3 WMMA GEMMs -> gated scatter ----------------
// 128 threads = 4 waves; 16 edges per wave.
__global__ __launch_bounds__(128) void edge_conv(
    const _Float16* __restrict__ xh, const int* __restrict__ esrc,
    const int* __restrict__ etgt,
    const _Float16* __restrict__ pWe1,   // packed, K=128 N=256
    const _Float16* __restrict__ pWe2,   // packed, K=256 N=128
    const _Float16* __restrict__ pWg,    // packed, K=128 N=128
    float* __restrict__ agg, int nEdges)
{
  __shared__ alignas(32) _Float16 Zs[4][16][HDIM];   // 16 KB
  __shared__ alignas(32) _Float16 Ts[4][16][EFEAT];  // 32 KB
  __shared__ int tg[4][16];

  const int tid   = threadIdx.x;
  const int wave  = tid >> 5;
  const int lane  = tid & 31;
  const int eBase = blockIdx.x * 64 + wave * 16;

  // gather z = xh[edge_sources] into LDS: pure f16 copy, 2 rows per iteration
  // (lane 0-15 -> first row halves, lane 16-31 -> second row halves)
  const int half = lane >> 4;          // 0/1: which of the two rows this lane serves
  const int col8 = (lane & 15) * 8;    // 8 f16 (16 B) per lane per row
#pragma unroll
  for (int it = 0; it < 8; ++it) {
    int r = it * 2 + half;
    int e = eBase + r;
    int src = esrc[e < nEdges ? e : 0];
    v8h hv = *(const v8h*)(xh + (size_t)src * HDIM + col8);
    *(v8h*)(&Zs[wave][r][col8]) = hv;
  }
  if (lane < 16) {
    int e = eBase + lane;
    tg[wave][lane] = (e < nEdges) ? etgt[e] : -1;
  }
  __syncthreads();

  const int arow = lane & 15;
  const int akb  = (lane >> 4) << 3;   // A-fragment K sub-offset (0 or 8)
  const int cn   = lane & 15;          // C/D: column
  const int cm0  = (lane >> 4) << 3;   // C/D: row base (0 or 8)
  const _Float16* zrow = &Zs[wave][arow][0];
  const _Float16* trow = &Ts[wave][arow][0];

  // Hoist Z A-fragments (K=128 -> 4 frags) into registers.
  v16h za[4];
#pragma unroll
  for (int kt = 0; kt < 4; ++kt) za[kt] = load_a_frag(zrow, kt * 32 + akb);

  // GEMM1: T = softplus(Z @ We1)    M=16 K=128 N=256
#pragma unroll 2
  for (int nt = 0; nt < 16; ++nt) {
    v8f acc = {};
#pragma unroll
    for (int kt = 0; kt < 4; ++kt) {
      v16h b = *(const v16h*)(pWe1 + ((size_t)(kt * 16 + nt) * 32 + lane) * 16);
      acc = wmma_f16(za[kt], b, acc);
    }
#pragma unroll
    for (int r = 0; r < 8; ++r)
      Ts[wave][cm0 + r][nt * 16 + cn] = (_Float16)sp(acc[r]);
  }
  __syncthreads();

  // Hoist T A-fragments (K=256 -> 8 frags) into registers.
  v16h ta[8];
#pragma unroll
  for (int kt = 0; kt < 8; ++kt) ta[kt] = load_a_frag(trow, kt * 32 + akb);

  // GEMM2: e = T @ We2 (M=16 K=256 N=128); GEMM3: gate = Z @ Wg (K=128 N=128)
#pragma unroll 2
  for (int nt = 0; nt < 8; ++nt) {
    v8f acc2 = {};
#pragma unroll
    for (int kt = 0; kt < 8; ++kt) {
      v16h b = *(const v16h*)(pWe2 + ((size_t)(kt * 8 + nt) * 32 + lane) * 16);
      acc2 = wmma_f16(ta[kt], b, acc2);
    }
    v8f accg = {};
#pragma unroll
    for (int kt = 0; kt < 4; ++kt) {
      v16h b = *(const v16h*)(pWg + ((size_t)(kt * 8 + nt) * 32 + lane) * 16);
      accg = wmma_f16(za[kt], b, accg);
    }
    const int n = nt * 16 + cn;
#pragma unroll
    for (int r = 0; r < 8; ++r) {
      int t = tg[wave][cm0 + r];
      if (t >= 0) {
        float mv = sigm(accg[r]) * sp(acc2[r]);
        atomicAdd(agg + (size_t)t * HDIM + n, mv);   // global_atomic_add_f32
      }
    }
  }
}

// ---------------- pooling: y[g] += x @ Wp (head divides by counts) ----------------
__global__ __launch_bounds__(128) void pool_kernel(
    const _Float16* __restrict__ xh, const int* __restrict__ gidx,
    const _Float16* __restrict__ pWp, float* __restrict__ y, int nNodes)
{
  __shared__ alignas(32) _Float16 Zs[4][16][HDIM];
  __shared__ int gof[4][16];

  const int tid   = threadIdx.x;
  const int wave  = tid >> 5;
  const int lane  = tid & 31;
  const int nBase = blockIdx.x * 64 + wave * 16;

  const int half = lane >> 4;
  const int col8 = (lane & 15) * 8;
#pragma unroll
  for (int it = 0; it < 8; ++it) {
    int r = it * 2 + half;
    int node = nBase + r;
    v8h hv = {};
    if (node < nNodes) hv = *(const v8h*)(xh + (size_t)node * HDIM + col8);
    *(v8h*)(&Zs[wave][r][col8]) = hv;
  }
  if (lane < 16) {
    int node = nBase + lane;
    gof[wave][lane] = (node < nNodes) ? gidx[node] : -1;
  }
  __syncthreads();

  const int arow = lane & 15;
  const int akb  = (lane >> 4) << 3;
  const int cn   = lane & 15;
  const int cm0  = (lane >> 4) << 3;
  const _Float16* zrow = &Zs[wave][arow][0];

  v16h za[4];
#pragma unroll
  for (int kt = 0; kt < 4; ++kt) za[kt] = load_a_frag(zrow, kt * 32 + akb);

#pragma unroll 2
  for (int nt = 0; nt < 8; ++nt) {
    v8f acc = {};
#pragma unroll
    for (int kt = 0; kt < 4; ++kt) {
      v16h b = *(const v16h*)(pWp + ((size_t)(kt * 8 + nt) * 32 + lane) * 16);
      acc = wmma_f16(za[kt], b, acc);
    }
    const int n = nt * 16 + cn;
#pragma unroll
    for (int r = 0; r < 8; ++r) {
      int g = gof[wave][cm0 + r];
      if (g >= 0) atomicAdd(y + (size_t)g * HDIM + n, acc[r]);
    }
  }
}

// ---------------- MLP head: 1 block (128 thr) per graph, f32 ----------------
__global__ __launch_bounds__(128) void head_kernel(
    const float* __restrict__ y, const float* __restrict__ counts,
    const float* __restrict__ Wfc1, const float* __restrict__ b1,
    const float* __restrict__ Wfc2, const float* __restrict__ b2,
    const float* __restrict__ Wr, const float* __restrict__ br,
    float* __restrict__ out)
{
  const int g = blockIdx.x, t = threadIdx.x;
  __shared__ float s0[128], s1[128];
  s0[t] = sp(y[(size_t)g * HDIM + t] * __builtin_amdgcn_rcpf(counts[g]));
  __syncthreads();
  float a = b1[t];
  for (int k = 0; k < HDIM; ++k) a += s0[k] * Wfc1[(size_t)k * 128 + t];
  s1[t] = sp(a);
  __syncthreads();
  a = b2[t];
  for (int k = 0; k < 128; ++k) a += s1[k] * Wfc2[(size_t)k * 128 + t];
  float gv = sp(a) * Wr[t];
  __syncthreads();
  s0[t] = gv;
  __syncthreads();
  for (int s = 64; s > 0; s >>= 1) {
    if (t < s) s0[t] += s0[t + s];
    __syncthreads();
  }
  if (t == 0) out[g] = s0[0] + br[0];
}

// ---------------- host launch ----------------
extern "C" void kernel_launch(void* const* d_in, const int* in_sizes, int n_in,
                              void* d_out, int out_size, void* d_ws, size_t ws_size,
                              hipStream_t stream) {
  (void)n_in; (void)out_size; (void)ws_size;
  const int*   nodes  = (const int*)d_in[0];
  const int*   esrc   = (const int*)d_in[1];
  const int*   etgt   = (const int*)d_in[2];
  const int*   gidx   = (const int*)d_in[3];
  const float* counts = (const float*)d_in[4];
  const float* emb    = (const float*)d_in[5];
  const float* Wg     = (const float*)d_in[6];
  const float* We1    = (const float*)d_in[7];
  const float* We2    = (const float*)d_in[8];
  const float* Wp     = (const float*)d_in[9];
  const float* Wfc1   = (const float*)d_in[10];
  const float* b1     = (const float*)d_in[11];
  const float* Wfc2   = (const float*)d_in[12];
  const float* b2     = (const float*)d_in[13];
  const float* Wr     = (const float*)d_in[14];
  const float* br     = (const float*)d_in[15];

  const int nNodes  = in_sizes[0];
  const int nEdges  = in_sizes[1];
  const int nGraphs = in_sizes[4];
  const int nConv   = in_sizes[6] / (HDIM * HDIM);   // 3

  // workspace carve (256-B aligned regions)
  char* p = (char*)d_ws;
  auto carve = [&](size_t bytes) {
    char* r = p; p += (bytes + 255) & ~(size_t)255; return r;
  };
  _Float16* xh   = (_Float16*)carve((size_t)nNodes  * HDIM * 2);   // f16 node state
  float*    agg  = (float*)   carve((size_t)nNodes  * HDIM * 4);
  float*    y    = (float*)   carve((size_t)nGraphs * HDIM * 4);
  _Float16* pWe1 = (_Float16*)carve((size_t)nConv * HDIM * EFEAT * 2);
  _Float16* pWe2 = (_Float16*)carve((size_t)nConv * EFEAT * HDIM * 2);
  _Float16* pWg  = (_Float16*)carve((size_t)nConv * HDIM * HDIM * 2);
  _Float16* pWp  = (_Float16*)carve((size_t)HDIM * HDIM * 2);

  // pack weights to f16 WMMA B-fragment layout
  for (int i = 0; i < nConv; ++i) {
    pack_b<<<(HDIM * EFEAT + 255) / 256, 256, 0, stream>>>(
        We1 + (size_t)i * HDIM * EFEAT, pWe1 + (size_t)i * HDIM * EFEAT, HDIM, EFEAT);
    pack_b<<<(EFEAT * HDIM + 255) / 256, 256, 0, stream>>>(
        We2 + (size_t)i * EFEAT * HDIM, pWe2 + (size_t)i * EFEAT * HDIM, EFEAT, HDIM);
    pack_b<<<(HDIM * HDIM + 255) / 256, 256, 0, stream>>>(
        Wg + (size_t)i * HDIM * HDIM, pWg + (size_t)i * HDIM * HDIM, HDIM, HDIM);
  }
  pack_b<<<(HDIM * HDIM + 255) / 256, 256, 0, stream>>>(Wp, pWp, HDIM, HDIM);

  // xh = (f16) embed[nodes]
  embed_gather<<<((size_t)nNodes * HDIM + 255) / 256, 256, 0, stream>>>(nodes, emb, xh, nNodes);

  // 3 gated convolutions
  for (int i = 0; i < nConv; ++i) {
    hipMemsetAsync(agg, 0, (size_t)nNodes * HDIM * 4, stream);
    edge_conv<<<(nEdges + 63) / 64, 128, 0, stream>>>(
        xh, esrc, etgt,
        pWe1 + (size_t)i * HDIM * EFEAT,
        pWe2 + (size_t)i * EFEAT * HDIM,
        pWg  + (size_t)i * HDIM * HDIM,
        agg, nEdges);
    residual<<<((size_t)nNodes * HDIM + 255) / 256, 256, 0, stream>>>(xh, agg, nNodes * HDIM);
  }

  // pooling + head
  hipMemsetAsync(y, 0, (size_t)nGraphs * HDIM * 4, stream);
  pool_kernel<<<(nNodes + 63) / 64, 128, 0, stream>>>(xh, gidx, pWp, y, nNodes);
  head_kernel<<<nGraphs, 128, 0, stream>>>(y, counts, Wfc1, b1, Wfc2, b2, Wr, br,
                                           (float*)d_out);
}
